// HardMixingLoss_5669356835815
// MI455X (gfx1250) — matrix operand order repaired and checked by reading, not compiled, not run on hardware
//
#include <hip/hip_runtime.h>
#include <hip/hip_bf16.h>
#include <math.h>

// Problem constants (from reference): N=2048, D=128, S=150, thr=204, tau=0.5, mixup=0.2
#define N_ROWS   2048
#define D_DIM    128
#define NP       4096        // pooled rows = 2N
#define THR      204
#define S_CNT    150
#define TWO_S    300
#define TOP_STRIDE 208       // padded stride for top-k index rows
#define LDS_STR  136         // half-element stride for LDS tiles (conflict padding, keeps 16B align)

typedef _Float16 half_t;
typedef __attribute__((ext_vector_type(4)))  _Float16 v4h;
typedef __attribute__((ext_vector_type(8)))  _Float16 v8h;
typedef __attribute__((ext_vector_type(16))) _Float16 v16h;
typedef __attribute__((ext_vector_type(8)))  float    v8f;

__device__ __forceinline__ float wave_sum(float v) {
#pragma unroll
  for (int off = 16; off > 0; off >>= 1) v += __shfl_xor(v, off, 32);
  return v;
}

// A fragment (16x32 f16, M x K). ISA 7.12.2: lanes 0-15 -> M=lane, K = k0+{0..7} then k0+{16..23};
// lanes 16-31 -> M=lane-16, K = k0+{8..15} then k0+{24..31}.
__device__ __forceinline__ v16h load_a_frag(const half_t* base, int stride, int k0, int lane) {
  const int m  = lane & 15;
  const int hi = (lane >> 4) & 1;
  const half_t* p = base + m * stride + k0 + hi * 8;
  v8h lo  = *(const v8h*)(p);        // 16B -> ds_load_b128
  v8h hi8 = *(const v8h*)(p + 16);
  v16h r;
#pragma unroll
  for (int i = 0; i < 8; ++i) { r[i] = lo[i]; r[8 + i] = hi8[i]; }
  return r;
}

// B fragment (32x16 f16, K x N), stored as rows-of-K per N: lane = N (mod 16),
// lanes 0-15 hold K = k0+{0..7} then k0+{16..23}, lanes 16-31 hold K = k0+{8..15} then k0+{24..31}.
__device__ __forceinline__ v16h load_b_frag(const half_t* base, int stride, int k0, int lane) {
  const int n  = lane & 15;
  const int hi = (lane >> 4) & 1;
  const half_t* p = base + n * stride + k0 + hi * 8;
  v8h lo  = *(const v8h*)(p);
  v8h hi8 = *(const v8h*)(p + 16);
  v16h r;
#pragma unroll
  for (int i = 0; i < 8; ++i) { r[i] = lo[i]; r[8 + i] = hi8[i]; }
  return r;
}

// ---------------- Kernel 1: row-normalize z1/z2 into f16 pool matrix Mh [4096][128] ----------
__global__ void k_normalize(const float* __restrict__ z1, const float* __restrict__ z2,
                            half_t* __restrict__ Mh) {
  const int wave = threadIdx.x >> 5, lane = threadIdx.x & 31;
  const int r = blockIdx.x * 4 + wave;               // grid = 1024 blocks * 4 waves
  const float* src = (r < N_ROWS) ? (z1 + (size_t)r * D_DIM)
                                  : (z2 + (size_t)(r - N_ROWS) * D_DIM);
  float4 x = reinterpret_cast<const float4*>(src)[lane];
  float ss = x.x * x.x + x.y * x.y + x.z * x.z + x.w * x.w;
  ss = wave_sum(ss);
  float inv = 1.0f / fmaxf(sqrtf(ss), 1e-12f);
  v4h o;
  o[0] = (half_t)(x.x * inv); o[1] = (half_t)(x.y * inv);
  o[2] = (half_t)(x.z * inv); o[3] = (half_t)(x.w * inv);
  *reinterpret_cast<v4h*>(Mh + (size_t)r * D_DIM + lane * 4) = o;
}

// ---------------- Kernel 1b: proj_w f32 -> f16 ----------------
__global__ void k_convw(const float* __restrict__ w, half_t* __restrict__ Wh) {
  int i = blockIdx.x * blockDim.x + threadIdx.x;
  if (i < D_DIM * D_DIM) Wh[i] = (half_t)w[i];
}

// ---------------- Kernel 2: P = Mh * Mh^T (4096x4096, K=128) via WMMA ----------------
// One block = 128x128 output tile; 8 waves, each wave owns a 16-row strip (8 col tiles).
__global__ void k_gemm_pool(const half_t* __restrict__ Mh, float* __restrict__ P) {
  extern __shared__ char smem[];
  half_t* At = (half_t*)smem;                 // 128 x LDS_STR
  half_t* Bt = At + 128 * LDS_STR;            // 128 x LDS_STR
  const int br = blockIdx.x >> 5, bc = blockIdx.x & 31;
  const int tid = threadIdx.x, wave = tid >> 5, lane = tid & 31;
  const size_t rowBase = (size_t)br * 128, colBase = (size_t)bc * 128;

  const v8h* srcA = (const v8h*)(Mh + rowBase * D_DIM);   // contiguous 32KB strip
  const v8h* srcB = (const v8h*)(Mh + colBase * D_DIM);
  __builtin_prefetch(srcB, 0, 3);                         // global_prefetch_b8
  for (int q = tid; q < 128 * 16; q += 256) {
    int rr = q >> 4, c8 = q & 15;
    *(v8h*)(At + rr * LDS_STR + c8 * 8) = srcA[q];
    *(v8h*)(Bt + rr * LDS_STR + c8 * 8) = srcB[q];
  }
  __syncthreads();

  v8f acc[8] = {};
  const half_t* aBase = At + (wave * 16) * LDS_STR;
#pragma unroll
  for (int ks = 0; ks < 4; ++ks) {
    v16h af = load_a_frag(aBase, LDS_STR, ks * 32, lane);
#pragma unroll
    for (int t = 0; t < 8; ++t) {
      v16h bf = load_b_frag(Bt + t * 16 * LDS_STR, LDS_STR, ks * 32, lane);
      acc[t] = __builtin_amdgcn_wmma_f32_16x16x32_f16(false, af, false, bf,
                                                      (short)0, acc[t], false, false);
    }
  }
  const int hi = (lane >> 4) & 1, nn = lane & 15;
#pragma unroll
  for (int t = 0; t < 8; ++t) {
#pragma unroll
    for (int r = 0; r < 8; ++r) {
      size_t row = rowBase + wave * 16 + r + hi * 8;
      size_t col = colBase + t * 16 + nn;
      P[row * NP + col] = acc[t][r];
    }
  }
}

// ---------------- Kernel 3: per-row exp-sum + top-204 indices (bitonic sort in LDS) ----------
__global__ void k_row_topk(const float* __restrict__ P, float* __restrict__ expsum,
                           int* __restrict__ topidx) {
  extern __shared__ char smem[];
  unsigned long long* keys = (unsigned long long*)smem;   // 4096 x u64
  float* red = (float*)(smem + (size_t)NP * 8);           // 256 floats
  const int tid = threadIdx.x;
  const int row = blockIdx.x;
  const float* prow = P + (size_t)row * NP;

  float es = 0.f;
  for (int i = tid; i < NP; i += 256) {
    float v = prow[i];
    es += __expf(2.0f * v);                               // 1/tau = 2
    unsigned b = __float_as_uint(v);
    b = (b & 0x80000000u) ? ~b : (b | 0x80000000u);       // order-preserving uint key
    // low 32 bits = ~idx so descending u64 sort breaks ties toward smaller idx (top_k stable)
    keys[i] = ((unsigned long long)b << 32) | (unsigned)(0xFFFFFFFFu - (unsigned)i);
  }
  red[tid] = es;
  __syncthreads();
  for (int s = 128; s > 0; s >>= 1) {
    if (tid < s) red[tid] += red[tid + s];
    __syncthreads();
  }
  if (tid == 0) expsum[row] = red[0];

  // bitonic sort, descending
  for (unsigned k = 2; k <= NP; k <<= 1) {
    for (unsigned j = k >> 1; j > 0; j >>= 1) {
      for (unsigned i = tid; i < NP; i += 256) {
        unsigned ixj = i ^ j;
        if (ixj > i) {
          unsigned long long a = keys[i], b = keys[ixj];
          bool up = ((i & k) == 0);
          if (up ? (a < b) : (a > b)) { keys[i] = b; keys[ixj] = a; }
        }
      }
      __syncthreads();
    }
  }
  if (tid < THR)
    topidx[(size_t)row * TOP_STRIDE + tid] =
        (int)(0xFFFFFFFFu - (unsigned)(keys[tid] & 0xFFFFFFFFull));
}

// ---------------- Kernel 4: gather + mixup + projection GEMM (WMMA) + normalize/dot/exp ------
// One block per (row n, view v). 150x128 @ 128x128 GEMM per block.
__global__ void k_mix_proj(const float* __restrict__ z1, const float* __restrict__ z2,
                           const float* __restrict__ proj_b,
                           const int* __restrict__ idx1, const int* __restrict__ idx2,
                           const half_t* __restrict__ Mh, const half_t* __restrict__ Wh,
                           const int* __restrict__ topidx, float* __restrict__ negm) {
  extern __shared__ char smem[];
  half_t* At  = (half_t*)smem;                          // 160 x LDS_STR (mix, f16)
  half_t* Bt  = At + 160 * LDS_STR;                     // 128 x LDS_STR (W,   f16)
  float*  hm  = (float*)(Bt + 128 * LDS_STR);           // 160 x 128 (proj out, f32)
  int*    gidx = (int*)(hm + 160 * D_DIM);              // 304 ints
  float*  red  = (float*)(gidx + 304);                  // 8 floats
  const int tid = threadIdx.x, wave = tid >> 5, lane = tid & 31;
  const int n = blockIdx.x & (N_ROWS - 1);
  const int v = blockIdx.x >> 11;
  const int rowp = v * N_ROWS + n;
  const int* idxArr = (v == 0) ? idx1 : idx2;
  const int* trow = topidx + (size_t)rowp * TOP_STRIDE;
  __builtin_prefetch(Mh + (size_t)rowp * D_DIM, 0, 3);

  for (int t = tid; t < TWO_S; t += 256)
    gidx[t] = trow[idxArr[(size_t)n * TWO_S + t]];

  const v8h* srcW = (const v8h*)Wh;
  for (int q = tid; q < 128 * 16; q += 256) {
    int rr = q >> 4, c8 = q & 15;
    *(v8h*)(Bt + rr * LDS_STR + c8 * 8) = srcW[q];
  }
  __syncthreads();

  // mix rows 0..149, zero-pad rows 150..159
  for (int e = tid; e < S_CNT * D_DIM; e += 256) {
    int s = e >> 7, d = e & 127;
    int ga = gidx[s], gb = gidx[S_CNT + s];
    const float* za = (ga < N_ROWS) ? (z1 + (size_t)ga * D_DIM)
                                    : (z2 + (size_t)(ga - N_ROWS) * D_DIM);
    const float* zb = (gb < N_ROWS) ? (z1 + (size_t)gb * D_DIM)
                                    : (z2 + (size_t)(gb - N_ROWS) * D_DIM);
    At[s * LDS_STR + d] = (half_t)(0.2f * za[d] + 0.8f * zb[d]);
  }
  for (int e = tid; e < 10 * D_DIM; e += 256) {
    int s = S_CNT + (e >> 7), d = e & 127;
    At[s * LDS_STR + d] = (half_t)0.0f;
  }
  __syncthreads();

  // wave w owns output column tile w (d = 16w..16w+15); hoist its 4 B k-fragments
  v16h bf[4];
#pragma unroll
  for (int ks = 0; ks < 4; ++ks)
    bf[ks] = load_b_frag(Bt + wave * 16 * LDS_STR, LDS_STR, ks * 32, lane);
  const int hi = (lane >> 4) & 1, nn = lane & 15;
  const float bb = proj_b[wave * 16 + nn];

  for (int rs = 0; rs < 10; ++rs) {
    v8f acc = {};
#pragma unroll
    for (int ks = 0; ks < 4; ++ks) {
      v16h af = load_a_frag(At + rs * 16 * LDS_STR, LDS_STR, ks * 32, lane);
      acc = __builtin_amdgcn_wmma_f32_16x16x32_f16(false, af, false, bf[ks],
                                                   (short)0, acc, false, false);
    }
#pragma unroll
    for (int r = 0; r < 8; ++r) {
      int srow = rs * 16 + r + hi * 8;
      hm[srow * D_DIM + wave * 16 + nn] = acc[r] + bb;
    }
  }
  __syncthreads();

  // per-s: exp(2 * dot(normalize(h), n_view)) accumulated over s
  const half_t* nh = Mh + (size_t)rowp * D_DIM;
  float partial = 0.f;
  for (int s = wave; s < S_CNT; s += 8) {
    const float* h = hm + s * D_DIM;
    float s2 = 0.f, dt = 0.f;
#pragma unroll
    for (int j = 0; j < 4; ++j) {
      float hv = h[lane * 4 + j];
      s2 += hv * hv;
      dt += hv * (float)nh[lane * 4 + j];
    }
    s2 = wave_sum(s2);
    dt = wave_sum(dt);
    if (lane == 0) partial += __expf(2.0f * dt / fmaxf(sqrtf(s2), 1e-12f));
  }
  if (lane == 0) red[wave] = partial;
  __syncthreads();
  if (tid == 0) {
    float tot = 0.f;
    for (int wv = 0; wv < 8; ++wv) tot += red[wv];
    negm[rowp] = tot;
  }
}

// ---------------- Kernel 5: final loss reduction ----------------
__global__ void k_loss(const float* __restrict__ P, const float* __restrict__ expsum,
                       const float* __restrict__ negm, float* __restrict__ out) {
  __shared__ float red[256];
  const int tid = threadIdx.x;
  float acc = 0.f;
  for (int i = tid; i < N_ROWS; i += 256) {
    float refl1 = P[(size_t)i * NP + i];
    float refl2 = P[(size_t)(N_ROWS + i) * NP + (N_ROWS + i)];
    float posl  = 2.0f * P[(size_t)i * NP + (N_ROWS + i)];   // log(pos) = s12_ii / tau
    float d1 = expsum[i] + negm[i] - refl1;
    float d2 = expsum[N_ROWS + i] + negm[N_ROWS + i] - refl2;
    acc += 0.5f * ((logf(d1) - posl) + (logf(d2) - posl));
  }
  red[tid] = acc;
  __syncthreads();
  for (int s = 128; s > 0; s >>= 1) {
    if (tid < s) red[tid] += red[tid + s];
    __syncthreads();
  }
  if (tid == 0) out[0] = red[0] / (float)N_ROWS;
}

extern "C" void kernel_launch(void* const* d_in, const int* in_sizes, int n_in,
                              void* d_out, int out_size, void* d_ws, size_t ws_size,
                              hipStream_t stream) {
  (void)in_sizes; (void)n_in; (void)out_size; (void)ws_size;
  const float* z1     = (const float*)d_in[0];
  const float* z2     = (const float*)d_in[1];
  const float* proj_w = (const float*)d_in[2];
  const float* proj_b = (const float*)d_in[3];
  const int*   idx1   = (const int*)d_in[4];
  const int*   idx2   = (const int*)d_in[5];

  char* ws = (char*)d_ws;
  size_t off = 0;
  float*  P      = (float*)(ws + off);  off += (size_t)NP * NP * sizeof(float);      // 64 MB
  half_t* Mh     = (half_t*)(ws + off); off += (size_t)NP * D_DIM * sizeof(half_t);  // 1 MB
  half_t* Wh     = (half_t*)(ws + off); off += (size_t)D_DIM * D_DIM * sizeof(half_t);
  float*  expsum = (float*)(ws + off);  off += (size_t)NP * sizeof(float);
  float*  negm   = (float*)(ws + off);  off += (size_t)NP * sizeof(float);
  int*    topidx = (int*)(ws + off);    off += (size_t)NP * TOP_STRIDE * sizeof(int);

  k_normalize<<<NP / 4, 128, 0, stream>>>(z1, z2, Mh);
  k_convw<<<(D_DIM * D_DIM + 255) / 256, 256, 0, stream>>>(proj_w, Wh);

  size_t gemmLds = (size_t)2 * 128 * LDS_STR * sizeof(half_t);
  k_gemm_pool<<<32 * 32, 256, gemmLds, stream>>>(Mh, P);

  size_t sortLds = (size_t)NP * 8 + 256 * sizeof(float);
  k_row_topk<<<NP, 256, sortLds, stream>>>(P, expsum, topidx);

  size_t mixLds = (size_t)(160 + 128) * LDS_STR * sizeof(half_t)
                + (size_t)160 * D_DIM * sizeof(float)
                + 304 * sizeof(int) + 8 * sizeof(float);
  k_mix_proj<<<2 * N_ROWS, 256, mixLds, stream>>>(z1, z2, proj_b, idx1, idx2,
                                                  Mh, Wh, topidx, negm);

  k_loss<<<1, 256, 0, stream>>>(P, expsum, negm, (float*)d_out);
}
